// StandardMoE_11089605558283
// MI455X (gfx1250) — compile-verified, matrix operand order checked by reading, not run on hardware
//
#include <hip/hip_runtime.h>
#include <hip/hip_bf16.h>

// ---------------------------------------------------------------------------
// MoE (N=8192, DIN=1024, DH=4096, DOUT=1024, E=8, K=2) for MI455X (gfx1250).
// Compute-bound (~950 FLOP/byte routed) -> bf16 WMMA path, f32 accumulate.
// bf16 weights (128MB) + bf16 x (16MB) fit the 192MB L2, so expert tiles
// re-stream B fragments from L2, not HBM. x-tile gather staged via the
// Tensor Data Mover (gather mode, 16 row indices) when available.
// ---------------------------------------------------------------------------

#define N_TOK 8192
#define DIN   1024
#define DH    4096
#define DOUT  1024
#define NEXP  8
#define TILE_M 16
#define AUX_EPS 1e-10f

#ifndef HAVE_TDM
#if defined(__has_builtin)
#if __has_builtin(__builtin_amdgcn_tensor_load_to_lds)
#define HAVE_TDM 1
#endif
#endif
#endif
#ifndef HAVE_TDM
#define HAVE_TDM 0
#endif

typedef __attribute__((ext_vector_type(8)))  __bf16 v8bf;
typedef __attribute__((ext_vector_type(16))) __bf16 v16bf;
typedef __attribute__((ext_vector_type(8)))  float  v8f;
typedef __attribute__((ext_vector_type(4)))  unsigned int u32x4;
typedef __attribute__((ext_vector_type(4)))  int i32x4;
typedef __attribute__((ext_vector_type(8)))  int i32x8;

__device__ __forceinline__ __bf16 f2bf(float f) {
  __hip_bfloat16 t = __float2bfloat16(f);
  __bf16 r;
  __builtin_memcpy(&r, &t, sizeof(r));
  return r;
}

__device__ __forceinline__ v8f splat8(float v) {
  v8f r;
#pragma unroll
  for (int i = 0; i < 8; ++i) r[i] = v;
  return r;
}

// Load a 16-lane-wide WMMA operand fragment (A or B) for 16x16x32 bf16.
// Per ISA layout: lanes 0-15 hold K 0..7 then 16..23; lanes 16-31 hold
// K 8..15 then 24..31. 'p' is already offset by row*stride + k0 + kh*8.
__device__ __forceinline__ v16bf frag_load(const __bf16* p) {
  v8bf lo = *(const v8bf*)(p);
  v8bf hi = *(const v8bf*)(p + 16);
  return __builtin_shufflevector(lo, hi, 0, 1, 2, 3, 4, 5, 6, 7,
                                         8, 9, 10, 11, 12, 13, 14, 15);
}

__device__ __forceinline__ v8f wmma_bf16(v16bf a, v16bf b, v8f c) {
  return __builtin_amdgcn_wmma_f32_16x16x32_bf16(
      /*neg_a=*/false, a, /*neg_b=*/false, b,
      /*c_mod=*/(short)0, c, /*reuse_a=*/false, /*reuse_b=*/false);
}

// ---------------------------------------------------------------------------
// 0) zero output region + expert counters
// ---------------------------------------------------------------------------
__global__ void zero_kernel(float* __restrict__ out, size_t n4,
                            int* __restrict__ counts) {
  size_t i = (size_t)blockIdx.x * blockDim.x + threadIdx.x;
  size_t stride = (size_t)gridDim.x * blockDim.x;
  float4 z = make_float4(0.f, 0.f, 0.f, 0.f);
  for (; i < n4; i += stride) ((float4*)out)[i] = z;
  if (blockIdx.x == 0 && threadIdx.x < NEXP) counts[threadIdx.x] = 0;
}

// ---------------------------------------------------------------------------
// 1) fp32 -> bf16 elementwise (x)
// ---------------------------------------------------------------------------
__global__ void cvt_kernel(const float* __restrict__ in,
                           __bf16* __restrict__ out, int n) {
  int i = blockIdx.x * blockDim.x + threadIdx.x;
  int stride = gridDim.x * blockDim.x;
  for (; i < n; i += stride) out[i] = f2bf(in[i]);
}

// ---------------------------------------------------------------------------
// 2) fp32 [E][R][C] -> bf16 transposed [E][C][R]; R,C multiples of 32
// ---------------------------------------------------------------------------
__global__ void transpose_kernel(const float* __restrict__ in,
                                 __bf16* __restrict__ out, int R, int C) {
  __shared__ float tile[32][33];
  int e = blockIdx.z;
  int c0 = blockIdx.x * 32, r0 = blockIdx.y * 32;
  const float* src = in + (size_t)e * R * C;
  __bf16* dst = out + (size_t)e * R * C;
  int tx = threadIdx.x, ty = threadIdx.y;  // 32 x 8
#pragma unroll
  for (int i = 0; i < 32; i += 8)
    tile[ty + i][tx] = src[(size_t)(r0 + ty + i) * C + (c0 + tx)];
  __syncthreads();
#pragma unroll
  for (int i = 0; i < 32; i += 8)
    dst[(size_t)(c0 + ty + i) * R + (r0 + tx)] = f2bf(tile[tx][ty + i]);
}

// ---------------------------------------------------------------------------
// 3) gating: one wave32 per token
// ---------------------------------------------------------------------------
__global__ void gate_kernel(const float* __restrict__ x,
                            const float* __restrict__ gw,
                            const float* __restrict__ gb,
                            float* __restrict__ probs_out,
                            float* __restrict__ block_prob,
                            int* __restrict__ counts,
                            int* __restrict__ btok,
                            float* __restrict__ bwgt) {
  __shared__ float wprob[8][NEXP];
  int wave = threadIdx.x >> 5;
  int lane = threadIdx.x & 31;
  int token = blockIdx.x * 8 + wave;

  float lg[NEXP];
#pragma unroll
  for (int e = 0; e < NEXP; ++e) lg[e] = 0.f;

  const float* xr = x + (size_t)token * DIN;
  for (int k = lane; k < DIN; k += 32) {
    float xv = xr[k];
    const float* g = gw + (size_t)k * NEXP;
#pragma unroll
    for (int e = 0; e < NEXP; ++e) lg[e] += xv * g[e];
  }
#pragma unroll
  for (int off = 16; off > 0; off >>= 1) {
#pragma unroll
    for (int e = 0; e < NEXP; ++e) lg[e] += __shfl_xor(lg[e], off, 32);
  }

  if (lane == 0) {
    float p[NEXP];
    float m = -3.4e38f;
#pragma unroll
    for (int e = 0; e < NEXP; ++e) {
      lg[e] += gb[e];
      m = fmaxf(m, lg[e]);
    }
    float s = 0.f;
#pragma unroll
    for (int e = 0; e < NEXP; ++e) { p[e] = __expf(lg[e] - m); s += p[e]; }
    float inv = 1.f / s;
#pragma unroll
    for (int e = 0; e < NEXP; ++e) {
      p[e] *= inv;
      probs_out[(size_t)token * NEXP + e] = p[e];
      wprob[wave][e] = p[e];
    }
    // top-2 (ties -> lowest index, matching lax.top_k)
    int i0 = 0; float v0 = p[0];
#pragma unroll
    for (int e = 1; e < NEXP; ++e) if (p[e] > v0) { v0 = p[e]; i0 = e; }
    int i1 = (i0 == 0) ? 1 : 0; float v1 = p[i1];
#pragma unroll
    for (int e = 0; e < NEXP; ++e)
      if (e != i0 && p[e] > v1) { v1 = p[e]; i1 = e; }
    float norm = 1.f / (v0 + v1 + AUX_EPS);
    int p0 = atomicAdd(&counts[i0], 1);
    btok[(size_t)i0 * N_TOK + p0] = token;
    bwgt[(size_t)i0 * N_TOK + p0] = v0 * norm;
    int p1 = atomicAdd(&counts[i1], 1);
    btok[(size_t)i1 * N_TOK + p1] = token;
    bwgt[(size_t)i1 * N_TOK + p1] = v1 * norm;
  }
  __syncthreads();
  if (threadIdx.x < NEXP) {
    float s = 0.f;
#pragma unroll
    for (int w = 0; w < 8; ++w) s += wprob[w][threadIdx.x];
    block_prob[(size_t)blockIdx.x * NEXP + threadIdx.x] = s;
  }
}

// ---------------------------------------------------------------------------
// 4) fused expert FFN: block = (expert, 16-token tile); 8 waves / block.
//    LDS: x tile (16x1024 bf16 = 32KB) + h tile (16x4096 bf16 = 128KB).
//    4 n-subtiles share each A fragment -> 4 independent WMMA chains.
// ---------------------------------------------------------------------------
__global__ void __launch_bounds__(256)
ffn_kernel(const __bf16* __restrict__ xb,
           const __bf16* __restrict__ w1t,   // [E][DH][DIN] bf16
           const __bf16* __restrict__ w2t,   // [E][DOUT][DH] bf16
           const float* __restrict__ b1,     // [E][DH]
           const float* __restrict__ b2,     // [E][DOUT]
           const int* __restrict__ btok,
           const float* __restrict__ bwgt,
           const int* __restrict__ counts,
           float* __restrict__ out) {
  extern __shared__ char smem[];
  __bf16* xs = (__bf16*)smem;                       // 16 * 1024 bf16
  __bf16* hs = (__bf16*)(smem + 32768);             // 16 * 4096 bf16
  int*    stok = (int*)(smem + 163840);             // 16 ints
  float*  swgt = (float*)(smem + 163904);           // 16 floats

  const int e = blockIdx.y;
  const int tileBase = blockIdx.x * TILE_M;
  const int cnt = counts[e];
  if (tileBase >= cnt) return;

  const int tid = threadIdx.x;
  if (tid < TILE_M) {
    int idx = tileBase + tid;
    if (idx < cnt) {
      stok[tid] = btok[(size_t)e * N_TOK + idx];
      swgt[tid] = bwgt[(size_t)e * N_TOK + idx];
    } else {
      stok[tid] = 0;       // pad with token 0,
      swgt[tid] = 0.f;     // zero combine weight -> contributes 0
    }
  }
  __syncthreads();

#if HAVE_TDM
  // ---- Tensor Data Mover, gather mode: 16 row indices (16-bit), one
  //      row = 1024 bf16 (2KB), written densely to LDS at xs. One wave
  //      issues the DMA; TENSORcnt tracks completion.
  if (tid < 32) {
    unsigned long long ga = (unsigned long long)(const void*)xb;
    u32x4 g0;
    g0[0] = 0x80000001u;     // count=1, is_restore=0, gather_mode=1, idx16
    g0[1] = 0u;              // lds_addr: xs at LDS offset 0
    g0[2] = (unsigned)(ga & 0xffffffffu);
    g0[3] = (unsigned)((ga >> 32) & 0x01ffffffu) | (2u << 30);  // type=2
    i32x8 g1;
    g1[0] = (int)(1u << 16);              // data_size=1 -> 2 bytes
    g1[1] = (int)((unsigned)DIN << 16);   // tensor_dim0 lo16 (=1024)
    g1[2] = (int)((unsigned)N_TOK << 16); // tensor_dim1 lo16 (=8192 rows)
    g1[3] = (int)((unsigned)DIN << 16);   // tile_dim0 = 1024
    g1[4] = 16;                           // tile_dim1 = #valid indices
    g1[5] = DIN;                          // tensor_dim0_stride lo32
    g1[6] = 0;
    g1[7] = 0;
    i32x4 g2, g3;
#pragma unroll
    for (int j = 0; j < 4; ++j) {
      unsigned a0 = (unsigned)__builtin_amdgcn_readfirstlane(stok[2 * j]);
      unsigned a1 = (unsigned)__builtin_amdgcn_readfirstlane(stok[2 * j + 1]);
      g2[j] = (int)(a0 | (a1 << 16));
      unsigned b0i = (unsigned)__builtin_amdgcn_readfirstlane(stok[8 + 2 * j]);
      unsigned b1i = (unsigned)__builtin_amdgcn_readfirstlane(stok[9 + 2 * j]);
      g3[j] = (int)(b0i | (b1i << 16));
    }
#if __clang_major__ >= 23
    i32x8 g4 = {};
    __builtin_amdgcn_tensor_load_to_lds(g0, g1, g2, g3, g4, 0);
#else
    __builtin_amdgcn_tensor_load_to_lds(g0, g1, g2, g3, 0);
#endif
    __builtin_amdgcn_s_wait_tensorcnt(0);
  }
#else
  // Fallback: cooperative gather copy, 2048 x 16B chunks.
  for (int c = tid; c < 2048; c += 256) {
    int row = c >> 7;
    int off = (c & 127) << 3;
    *(uint4*)(xs + (size_t)row * DIN + off) =
        *(const uint4*)(xb + (size_t)stok[row] * DIN + off);
  }
#endif
  __syncthreads();

  const int wave = tid >> 5;
  const int lane = tid & 31;
  const int r15 = lane & 15;
  const int kh = (lane >> 4) & 1;

  // ---- GEMM1: h[16,4096] = relu(x[16,1024] @ w1[e] + b1); wave w owns
  //      columns [w*512, w*512+512), 4 subtiles per A-fragment.
  {
    const int nbase = wave * 512;
    for (int g = 0; g < 8; ++g) {
      const int n0 = nbase + g * 64;
      const float* bb = b1 + (size_t)e * DH + n0 + r15;
      v8f acc0 = splat8(bb[0]);
      v8f acc1 = splat8(bb[16]);
      v8f acc2 = splat8(bb[32]);
      v8f acc3 = splat8(bb[48]);
      const __bf16* aB  = xs + (size_t)r15 * DIN + kh * 8;
      const __bf16* bB0 = w1t + ((size_t)e * DH + n0 + r15) * DIN + kh * 8;
      const __bf16* bB1 = bB0 + (size_t)16 * DIN;
      const __bf16* bB2 = bB0 + (size_t)32 * DIN;
      const __bf16* bB3 = bB0 + (size_t)48 * DIN;
#pragma unroll 2
      for (int k0 = 0; k0 < DIN; k0 += 32) {
        v16bf a = frag_load(aB + k0);
        __builtin_prefetch(bB0 + k0 + 256, 0, 3);
        acc0 = wmma_bf16(a, frag_load(bB0 + k0), acc0);
        acc1 = wmma_bf16(a, frag_load(bB1 + k0), acc1);
        acc2 = wmma_bf16(a, frag_load(bB2 + k0), acc2);
        acc3 = wmma_bf16(a, frag_load(bB3 + k0), acc3);
      }
#pragma unroll
      for (int r = 0; r < 8; ++r) {
        __bf16* hrow = hs + (size_t)(r + kh * 8) * DH + r15;
        hrow[n0]      = f2bf(fmaxf(acc0[r], 0.f));
        hrow[n0 + 16] = f2bf(fmaxf(acc1[r], 0.f));
        hrow[n0 + 32] = f2bf(fmaxf(acc2[r], 0.f));
        hrow[n0 + 48] = f2bf(fmaxf(acc3[r], 0.f));
      }
    }
  }
  __syncthreads();

  // ---- GEMM2: y[16,1024] = h[16,4096] @ w2[e] + b2; scale by gate weight
  //      and atomically accumulate (exactly 2 adds per output element).
  {
    const int nbase = wave * 128;
    for (int g = 0; g < 2; ++g) {
      const int n0 = nbase + g * 64;
      const float* bb = b2 + (size_t)e * DOUT + n0 + r15;
      v8f acc0 = splat8(bb[0]);
      v8f acc1 = splat8(bb[16]);
      v8f acc2 = splat8(bb[32]);
      v8f acc3 = splat8(bb[48]);
      const __bf16* aB  = hs + (size_t)r15 * DH + kh * 8;
      const __bf16* bB0 = w2t + ((size_t)e * DOUT + n0 + r15) * DH + kh * 8;
      const __bf16* bB1 = bB0 + (size_t)16 * DH;
      const __bf16* bB2 = bB0 + (size_t)32 * DH;
      const __bf16* bB3 = bB0 + (size_t)48 * DH;
#pragma unroll 2
      for (int k0 = 0; k0 < DH; k0 += 32) {
        v16bf a = frag_load(aB + k0);
        __builtin_prefetch(bB0 + k0 + 256, 0, 3);
        acc0 = wmma_bf16(a, frag_load(bB0 + k0), acc0);
        acc1 = wmma_bf16(a, frag_load(bB1 + k0), acc1);
        acc2 = wmma_bf16(a, frag_load(bB2 + k0), acc2);
        acc3 = wmma_bf16(a, frag_load(bB3 + k0), acc3);
      }
#pragma unroll
      for (int r = 0; r < 8; ++r) {
        int m = r + kh * 8;
        float wq = swgt[m];
        float* orow = out + (size_t)stok[m] * DOUT + r15;
        atomicAdd(&orow[n0],      wq * acc0[r]);
        atomicAdd(&orow[n0 + 16], wq * acc1[r]);
        atomicAdd(&orow[n0 + 32], wq * acc2[r]);
        atomicAdd(&orow[n0 + 48], wq * acc3[r]);
      }
    }
  }
}

// ---------------------------------------------------------------------------
// 5) aux loss: deterministic serial reduction of per-block prob partials
// ---------------------------------------------------------------------------
__global__ void aux_kernel(const float* __restrict__ block_prob, int nblocks,
                           float* __restrict__ aux_out) {
  __shared__ float mp[NEXP];
  if (threadIdx.x < NEXP) {
    float s = 0.f;
    for (int b = 0; b < nblocks; ++b)
      s += block_prob[(size_t)b * NEXP + threadIdx.x];
    mp[threadIdx.x] = s / (float)N_TOK;
  }
  __syncthreads();
  if (threadIdx.x == 0) {
    float a = 0.f;
#pragma unroll
    for (int e = 0; e < NEXP; ++e)
      a += mp[e] * logf(mp[e] * (float)NEXP + AUX_EPS);
    *aux_out = a;
  }
}

// ---------------------------------------------------------------------------
// launch
// ---------------------------------------------------------------------------
extern "C" void kernel_launch(void* const* d_in, const int* in_sizes, int n_in,
                              void* d_out, int out_size, void* d_ws,
                              size_t ws_size, hipStream_t stream) {
  const float* x  = (const float*)d_in[0];
  const float* gw = (const float*)d_in[1];
  const float* gb = (const float*)d_in[2];
  const float* w1 = (const float*)d_in[3];
  const float* b1 = (const float*)d_in[4];
  const float* w2 = (const float*)d_in[5];
  const float* b2 = (const float*)d_in[6];

  float* out   = (float*)d_out;                       // [N, DOUT]
  float* aux   = out + (size_t)N_TOK * DOUT;          // scalar
  float* probs = aux + 1;                             // [N, E]

  // workspace layout (bytes)
  char* ws = (char*)d_ws;
  const size_t OFF_W1T = 0;                           // 64 MB bf16
  const size_t OFF_W2T = OFF_W1T + (size_t)NEXP * DH * DIN * 2;   // 64 MB
  const size_t OFF_XB  = OFF_W2T + (size_t)NEXP * DOUT * DH * 2;  // 16 MB
  const size_t OFF_BT  = OFF_XB + (size_t)N_TOK * DIN * 2;        // 256 KB
  const size_t OFF_BW  = OFF_BT + (size_t)NEXP * N_TOK * 4;       // 256 KB
  const size_t OFF_CNT = OFF_BW + (size_t)NEXP * N_TOK * 4;       // 32 B
  const size_t OFF_BP  = OFF_CNT + 256;                           // 32 KB

  __bf16* w1t = (__bf16*)(ws + OFF_W1T);
  __bf16* w2t = (__bf16*)(ws + OFF_W2T);
  __bf16* xb  = (__bf16*)(ws + OFF_XB);
  int*    btok   = (int*)(ws + OFF_BT);
  float*  bwgt   = (float*)(ws + OFF_BW);
  int*    counts = (int*)(ws + OFF_CNT);
  float*  bprob  = (float*)(ws + OFF_BP);

  // 0) zero output + counters
  zero_kernel<<<2048, 256, 0, stream>>>(out, (size_t)N_TOK * DOUT / 4, counts);
  // 1) bf16 conversions (one-time; everything fits L2 afterwards)
  cvt_kernel<<<4096, 256, 0, stream>>>(x, xb, N_TOK * DIN);
  transpose_kernel<<<dim3(DH / 32, DIN / 32, NEXP), dim3(32, 8), 0, stream>>>(
      w1, w1t, DIN, DH);
  transpose_kernel<<<dim3(DOUT / 32, DH / 32, NEXP), dim3(32, 8), 0, stream>>>(
      w2, w2t, DH, DOUT);
  // 2) gating + routing
  gate_kernel<<<N_TOK / 8, 256, 0, stream>>>(x, gw, gb, probs, bprob, counts,
                                             btok, bwgt);
  // 3) fused expert FFN (WMMA + TDM gather staging)
  const size_t smem_bytes = 163968;  // 32KB x + 128KB h + tok/wgt arrays
  ffn_kernel<<<dim3(N_TOK / TILE_M, NEXP), 256, smem_bytes, stream>>>(
      xb, w1t, w2t, b1, b2, btok, bwgt, counts, out);
  // 4) aux loss
  aux_kernel<<<1, 32, 0, stream>>>(bprob, N_TOK / 8, aux);
}
